// PIGNN_61521111547997
// MI455X (gfx1250) — compile-verified, matrix operand order checked by reading, not run on hardware
//
#include <hip/hip_runtime.h>
#include <hip/hip_bf16.h>

#define NLAYERS 5
#define HD 128
#define MD 256
#define BGR 64
#define EPSBN 1e-5f

typedef __attribute__((ext_vector_type(16))) __bf16 v16bf;
typedef __attribute__((ext_vector_type(8)))  float  v8f;

union BFrag { unsigned short s[16]; unsigned int u[8]; uint4 q[2]; v16bf v; };

// ---------------------------------------------------------------------------
// f32 -> bf16 (RNE), packed pair. The clang builtin is not declared for
// gfx1250 on this toolchain, but CDNA5 has the BF16 datapath (BF16 WMMA,
// V_PK_*_BF16), so emit the hardware packed convert directly.
// ---------------------------------------------------------------------------
__device__ __forceinline__ unsigned int pk2bf(float a, float b) {
  unsigned int r;
  asm("v_cvt_pk_bf16_f32 %0, %1, %2" : "=v"(r) : "v"(a), "v"(b));
  return r;
}
__device__ __forceinline__ unsigned short f2bf(float f) {
  return (unsigned short)pk2bf(f, 0.f);
}

__device__ __forceinline__ float silu_f(float v) { return v / (1.f + __expf(-v)); }

// ---------------------------------------------------------------------------
// Weight repack: [128,128] f32 row-major (k,n) -> per-wave-fragment bf16 layout.
// Fragment mapping (16-bit A/B layout, 16x16x32): lane<16 half=0, lanes>=16 half=1;
// element e<8 -> K=kb+half*8+e, e>=8 -> K=kb+16+half*8+(e-8); lane&15 -> column n.
// dst index = ((nt*4+kc)*32 + lane)*16 + e  (each lane's 16 bf16 contiguous)
// ---------------------------------------------------------------------------
__global__ void pack_w(const float* __restrict__ w, unsigned short* __restrict__ dst) {
  int tid = blockIdx.x * blockDim.x + threadIdx.x;
  if (tid >= 8 * 4 * 32 * 16) return;
  int e    = tid & 15;
  int lane = (tid >> 4) & 31;
  int kc   = (tid >> 9) & 3;
  int nt   = tid >> 11;
  int n    = nt * 16 + (lane & 15);
  int half = lane >> 4;
  int k    = kc * 32 + (e < 8 ? half * 8 + e : 16 + half * 8 + (e - 8));
  dst[tid] = f2bf(w[k * HD + n]);
}

// ---------------------------------------------------------------------------
// Block LayerNorm helper (dim == blockDim.x, power of 2)
// ---------------------------------------------------------------------------
__device__ __forceinline__ float ln_block(float acc, float* red, int t, int dim,
                                          const float* g, const float* beta) {
  red[t] = acc; __syncthreads();
  for (int s = dim >> 1; s > 0; s >>= 1) { if (t < s) red[t] += red[t + s]; __syncthreads(); }
  float mu = red[0] / dim; __syncthreads();
  float d = acc - mu;
  red[t] = d * d; __syncthreads();
  for (int s = dim >> 1; s > 0; s >>= 1) { if (t < s) red[t] += red[t + s]; __syncthreads(); }
  float var = red[0] / dim; __syncthreads();
  return d * rsqrtf(var + EPSBN) * g[t] + beta[t];
}

// ---------------------------------------------------------------------------
// Node encoder: h = agg = SiLU(LN(x[N,16] @ W[16,128] + b)).  32 rows/block.
// Block 0 also zeroes the BN stats accumulator for layer 0.
// ---------------------------------------------------------------------------
__global__ void encode_nodes(const float* __restrict__ x, const float* __restrict__ w,
                             const float* __restrict__ b, const float* __restrict__ g,
                             const float* __restrict__ beta, float* __restrict__ h,
                             float* __restrict__ agg, float* __restrict__ stats,
                             int nrows) {
  __shared__ float sw[16 * HD];
  __shared__ float sx[16];
  __shared__ float red[HD];
  int t = threadIdx.x;                       // 128 threads
  if (blockIdx.x == 0) { stats[t] = 0.f; stats[HD + t] = 0.f; }
  for (int i = t; i < 16 * HD; i += HD) sw[i] = w[i];
  __syncthreads();
  int row0 = blockIdx.x * 32;
  for (int r = 0; r < 32; ++r) {
    int row = row0 + r;
    if (row >= nrows) break;                 // uniform across block
    if (t < 16) sx[t] = x[row * 16 + t];
    __syncthreads();
    float acc = b[t];
#pragma unroll
    for (int k = 0; k < 16; ++k) acc += sx[k] * sw[k * HD + t];
    float vv = silu_f(ln_block(acc, red, t, HD, g, beta));
    h[(size_t)row * HD + t] = vv;
    agg[(size_t)row * HD + t] = vv;
  }
}

// One wave per edge: 32 lanes x float4 = full 128-wide row, coalesced + L2 atomics
__global__ void scatter_edges(const float* __restrict__ h, float* __restrict__ agg,
                              const int* __restrict__ src, const int* __restrict__ dst,
                              int E) {
  long long tt = blockIdx.x * (long long)blockDim.x + threadIdx.x;
  int e = (int)(tt >> 5);
  if (e >= E) return;
  int lane = (int)(tt & 31);
  int s = src[e], d = dst[e];
  float4 v = *(const float4*)(h + (size_t)s * HD + lane * 4);
  float* o = agg + (size_t)d * HD + lane * 4;
  atomicAdd(o + 0, v.x); atomicAdd(o + 1, v.y);
  atomicAdd(o + 2, v.z); atomicAdd(o + 3, v.w);
}

// ---------------------------------------------------------------------------
// WMMA GEMM: C[nrows,128] = A[nrows,128] @ W(packed bf16) + bias
// A_BF16: A stored as bf16 (else f32, converted with packed cvt)
// EPI==0: SiLU then bf16 out (lin1)
// EPI==1: f32 out (lin2) + fused BN column statistics (LDS reduce -> global atomics)
// One wave = 16 rows x 128 cols, K=128 in 4 chunks of 32.
// ---------------------------------------------------------------------------
template<bool A_BF16, int EPI>
__global__ __launch_bounds__(256) void gemm_h(const void* __restrict__ Aptr,
                                              const unsigned short* __restrict__ wp,
                                              const float* __restrict__ bias,
                                              void* __restrict__ Cptr,
                                              float* __restrict__ stats, int nrows) {
  __shared__ float ssum[HD];
  __shared__ float ssq[HD];
  int wid  = threadIdx.x >> 5;
  int lane = threadIdx.x & 31;
  int tile = blockIdx.x * 8 + wid;
  int m0   = tile * 16;
  bool active = (m0 < nrows);                // wave-uniform
  if (EPI == 1) {
    if (threadIdx.x < HD) { ssum[threadIdx.x] = 0.f; ssq[threadIdx.x] = 0.f; }
    __syncthreads();
  }
  if (active) {
    int half = lane >> 4;
    int arow = m0 + (lane & 15);
    if (arow >= nrows) arow = nrows - 1;     // clamp loads (stores guarded)

    v8f acc[8] = {};
#pragma unroll
    for (int kc = 0; kc < 4; ++kc) {
      BFrag a;
      size_t base = (size_t)arow * HD + kc * 32 + half * 8;
      if (A_BF16) {
        const unsigned short* A = (const unsigned short*)Aptr;
        a.q[0] = *(const uint4*)(A + base);
        a.q[1] = *(const uint4*)(A + base + 16);
      } else {
        const float* A = (const float*)Aptr;
        float4 f0 = *(const float4*)(A + base);
        float4 f1 = *(const float4*)(A + base + 4);
        float4 f2 = *(const float4*)(A + base + 16);
        float4 f3 = *(const float4*)(A + base + 20);
        a.u[0] = pk2bf(f0.x, f0.y); a.u[1] = pk2bf(f0.z, f0.w);
        a.u[2] = pk2bf(f1.x, f1.y); a.u[3] = pk2bf(f1.z, f1.w);
        a.u[4] = pk2bf(f2.x, f2.y); a.u[5] = pk2bf(f2.z, f2.w);
        a.u[6] = pk2bf(f3.x, f3.y); a.u[7] = pk2bf(f3.z, f3.w);
      }
#pragma unroll
      for (int nt = 0; nt < 8; ++nt) {
        BFrag bq;
        const unsigned short* wsrc = wp + (((nt * 4 + kc) * 32) + lane) * 16;
        bq.q[0] = *(const uint4*)(wsrc);
        bq.q[1] = *(const uint4*)(wsrc + 8);
        acc[nt] = __builtin_amdgcn_wmma_f32_16x16x32_bf16(
            false, a.v, false, bq.v, (short)0, acc[nt], false, false);
      }
    }
    int half8 = half * 8;
    int colbase = lane & 15;
#pragma unroll
    for (int nt = 0; nt < 8; ++nt) {
      int col = nt * 16 + colbase;
      float bb = bias[col];
      float cs = 0.f, cq = 0.f;
#pragma unroll
      for (int r = 0; r < 8; ++r) {
        int row = m0 + half8 + r;
        if (row >= nrows) continue;
        float v = acc[nt][r] + bb;
        if (EPI == 0) {
          ((unsigned short*)Cptr)[(size_t)row * HD + col] = f2bf(silu_f(v));
        } else {
          ((float*)Cptr)[(size_t)row * HD + col] = v;
          cs += v; cq += v * v;
        }
      }
      if (EPI == 1) {
        atomicAdd(&ssum[col], cs);
        atomicAdd(&ssq[col], cq);
      }
    }
  }
  if (EPI == 1) {
    __syncthreads();
    if (threadIdx.x < HD) {
      atomicAdd(stats + threadIdx.x,      ssum[threadIdx.x]);
      atomicAdd(stats + HD + threadIdx.x, ssq[threadIdx.x]);
    }
  }
}

// Compute per-channel scale/shift from accumulated stats, then re-zero stats
// for the next layer.
__global__ void bn_finalize(float* __restrict__ stats, const float* __restrict__ g,
                            const float* __restrict__ b, float* __restrict__ ssc,
                            int nrows) {
  int t = threadIdx.x;
  if (t >= HD) return;
  float mu  = stats[t] / nrows;
  float var = stats[HD + t] / nrows - mu * mu;
  float sc  = g[t] * rsqrtf(var + EPSBN);
  ssc[t]      = sc;
  ssc[HD + t] = b[t] - mu * sc;
  stats[t] = 0.f;
  stats[HD + t] = 0.f;
}

// h = agg = SiLU(t*scale + shift)   (t aliases agg; next layer's self term
// is pre-seeded so no copy kernel is needed)
__global__ void bn_apply(float* __restrict__ ha, const float* __restrict__ ssc,
                         float* __restrict__ h, size_t n4) {
  size_t i = blockIdx.x * (size_t)blockDim.x + threadIdx.x;
  if (i >= n4) return;
  int c = (int)(i & (HD / 4 - 1)) * 4;
  float4 v = ((const float4*)ha)[i];
  float4 o;
  o.x = silu_f(v.x * ssc[c + 0] + ssc[HD + c + 0]);
  o.y = silu_f(v.y * ssc[c + 1] + ssc[HD + c + 1]);
  o.z = silu_f(v.z * ssc[c + 2] + ssc[HD + c + 2]);
  o.w = silu_f(v.w * ssc[c + 3] + ssc[HD + c + 3]);
  ((float4*)h)[i] = o;
  ((float4*)ha)[i] = o;
}

__global__ void pool_zero(float* hg, float* counts) {
  int i = blockIdx.x * blockDim.x + threadIdx.x;
  if (i < BGR * HD) hg[i] = 0.f;
  if (i < BGR) counts[i] = 0.f;
}

__global__ void pool_scatter(const float* __restrict__ h, const int* __restrict__ batch,
                             float* __restrict__ hg, float* __restrict__ counts,
                             int nrows) {
  long long tt = blockIdx.x * (long long)blockDim.x + threadIdx.x;
  int i = (int)(tt >> 5);
  if (i >= nrows) return;
  int lane = (int)(tt & 31);
  int bg = batch[i];
  float4 v = *(const float4*)(h + (size_t)i * HD + lane * 4);
  float* o = hg + (size_t)bg * HD + lane * 4;
  atomicAdd(o + 0, v.x); atomicAdd(o + 1, v.y);
  atomicAdd(o + 2, v.z); atomicAdd(o + 3, v.w);
  if (lane == 0) atomicAdd(counts + bg, 1.f);
}

__global__ void pool_finalize(float* hg, const float* counts) {
  int i = blockIdx.x * blockDim.x + threadIdx.x;
  if (i >= BGR * HD) return;
  hg[i] /= fmaxf(counts[i / HD], 1.f);
}

// phys encoder: SiLU(LN(u[64,8] @ W[8,128] + b))
__global__ void phys_enc(const float* __restrict__ u, const float* __restrict__ w,
                         const float* __restrict__ b, const float* __restrict__ g,
                         const float* __restrict__ beta, float* __restrict__ out) {
  __shared__ float su[8];
  __shared__ float red[HD];
  int t = threadIdx.x;                       // 128
  int row = blockIdx.x;
  if (t < 8) su[t] = u[row * 8 + t];
  __syncthreads();
  float acc = b[t];
#pragma unroll
  for (int k = 0; k < 8; ++k) acc += su[k] * w[k * HD + t];
  out[row * HD + t] = silu_f(ln_block(acc, red, t, HD, g, beta));
}

// fusion: SiLU(LN(concat(in0,in1)[64,256] @ W[256,256] + b))
__global__ void fusion(const float* __restrict__ in0, int stride0,
                       const float* __restrict__ in1, int stride1,
                       const float* __restrict__ w, const float* __restrict__ b,
                       const float* __restrict__ g, const float* __restrict__ beta,
                       float* __restrict__ out) {
  __shared__ float sin0[MD];
  __shared__ float red[MD];
  int t = threadIdx.x;                       // 256
  int row = blockIdx.x;
  sin0[t] = (t < HD) ? in0[row * stride0 + t] : in1[row * stride1 + (t - HD)];
  __syncthreads();
  float acc = b[t];
#pragma unroll 8
  for (int k = 0; k < MD; ++k) acc += sin0[k] * w[k * MD + t];
  out[row * MD + t] = silu_f(ln_block(acc, red, t, MD, g, beta));
}

// heads: theta[64] at out[0..63], z[64,128] at out[64..]
__global__ void heads(const float* __restrict__ hf, const float* __restrict__ wt,
                      const float* __restrict__ bt, const float* __restrict__ wz,
                      const float* __restrict__ bz, float* __restrict__ out) {
  __shared__ float sh[MD];
  int t = threadIdx.x;                       // 256
  int row = blockIdx.x;
  sh[t] = hf[row * MD + t];
  __syncthreads();
  if (t < HD) {
    float acc = bz[t];
#pragma unroll 8
    for (int k = 0; k < MD; ++k) acc += sh[k] * wz[k * HD + t];
    out[BGR + row * HD + t] = acc;
  }
  if (t == 0) {
    float acc = bt[0];
    for (int k = 0; k < MD; ++k) acc += sh[k] * wt[k];
    out[row] = acc;
  }
}

// ---------------------------------------------------------------------------
// Input leaf order = jax.tree_util (dict keys sorted alphabetically):
// 0:batch 1:edge_index
// 2+6l: conv[l]: bn_b, bn_g, lin1.b, lin1.w, lin2.b, lin2.w
// 32: enc.lin.b, enc.lin.w, enc.ln_b, enc.ln_g
// 36: fus1.lin.b, .lin.w, .ln_b, .ln_g   40: fus2...
// 44: phys.lin.b, .lin.w, .ln_b, .ln_g
// 48: theta.b, theta.w   50: z.b, z.w    52: u    53: x
// ---------------------------------------------------------------------------
extern "C" void kernel_launch(void* const* d_in, const int* in_sizes, int n_in,
                              void* d_out, int out_size, void* d_ws, size_t ws_size,
                              hipStream_t stream) {
  const int* batch = (const int*)d_in[0];
  const int* eidx  = (const int*)d_in[1];
  const int  N = in_sizes[0];
  const int  E = in_sizes[1] / 2;
  const int* src = eidx;
  const int* dst = eidx + E;

  const float *bn_b[NLAYERS], *bn_g[NLAYERS], *l1b[NLAYERS], *l1w[NLAYERS],
              *l2b[NLAYERS], *l2w[NLAYERS];
  for (int l = 0; l < NLAYERS; ++l) {
    bn_b[l] = (const float*)d_in[2 + 6 * l + 0];
    bn_g[l] = (const float*)d_in[2 + 6 * l + 1];
    l1b[l]  = (const float*)d_in[2 + 6 * l + 2];
    l1w[l]  = (const float*)d_in[2 + 6 * l + 3];
    l2b[l]  = (const float*)d_in[2 + 6 * l + 4];
    l2w[l]  = (const float*)d_in[2 + 6 * l + 5];
  }
  int p = 2 + 6 * NLAYERS;  // 32
  const float* enc_b   = (const float*)d_in[p + 0];
  const float* enc_w   = (const float*)d_in[p + 1];
  const float* enc_lnb = (const float*)d_in[p + 2];
  const float* enc_lng = (const float*)d_in[p + 3];
  const float* f1_b    = (const float*)d_in[p + 4];
  const float* f1_w    = (const float*)d_in[p + 5];
  const float* f1_lnb  = (const float*)d_in[p + 6];
  const float* f1_lng  = (const float*)d_in[p + 7];
  const float* f2_b    = (const float*)d_in[p + 8];
  const float* f2_w    = (const float*)d_in[p + 9];
  const float* f2_lnb  = (const float*)d_in[p + 10];
  const float* f2_lng  = (const float*)d_in[p + 11];
  const float* ph_b    = (const float*)d_in[p + 12];
  const float* ph_w    = (const float*)d_in[p + 13];
  const float* ph_lnb  = (const float*)d_in[p + 14];
  const float* ph_lng  = (const float*)d_in[p + 15];
  const float* th_b    = (const float*)d_in[p + 16];
  const float* th_w    = (const float*)d_in[p + 17];
  const float* z_b     = (const float*)d_in[p + 18];
  const float* z_w     = (const float*)d_in[p + 19];
  const float* u       = (const float*)d_in[p + 20];
  const float* x       = (const float*)d_in[p + 21];

  // workspace carve-out (256B aligned)
  char* ws = (char*)d_ws;
  size_t off = 0;
  auto carve = [&](size_t bytes) {
    void* ptr = ws + off;
    off += (bytes + 255) & ~(size_t)255;
    return ptr;
  };
  float*          h      = (float*)carve((size_t)N * HD * 4);
  float*          agg    = (float*)carve((size_t)N * HD * 4);   // also lin2 output
  unsigned short* t1     = (unsigned short*)carve((size_t)N * HD * 2);
  unsigned short* wpk    = (unsigned short*)carve((size_t)NLAYERS * 2 * HD * HD * 2);
  float*          stats  = (float*)carve(256 * 4);
  float*          ssc    = (float*)carve(256 * 4);
  float*          hg     = (float*)carve(BGR * HD * 4);
  float*          counts = (float*)carve(BGR * 4);
  float*          hphys  = (float*)carve(BGR * HD * 4);
  float*          hf1    = (float*)carve(BGR * MD * 4);
  float*          hf2    = (float*)carve(BGR * MD * 4);
  (void)ws_size; (void)n_in; (void)out_size;

  // per-launch weight repack to bf16 fragment layout
  for (int l = 0; l < NLAYERS; ++l) {
    pack_w<<<64, 256, 0, stream>>>(l1w[l], wpk + (size_t)(2 * l) * HD * HD);
    pack_w<<<64, 256, 0, stream>>>(l2w[l], wpk + (size_t)(2 * l + 1) * HD * HD);
  }

  encode_nodes<<<(N + 31) / 32, 128, 0, stream>>>(x, enc_w, enc_b, enc_lng, enc_lnb,
                                                  h, agg, stats, N);

  const size_t n4 = (size_t)N * HD / 4;
  const int elBlocks   = (int)((n4 + 255) / 256);
  const int ewBlocks   = (int)(((long long)E * 32 + 255) / 256);
  const int tiles      = (N + 15) / 16;
  const int gemmBlocks = (tiles + 7) / 8;

  for (int l = 0; l < NLAYERS; ++l) {
    scatter_edges<<<ewBlocks, 256, 0, stream>>>(h, agg, src, dst, E);
    gemm_h<false, 0><<<gemmBlocks, 256, 0, stream>>>(
        agg, wpk + (size_t)(2 * l) * HD * HD, l1b[l], t1, nullptr, N);
    gemm_h<true, 1><<<gemmBlocks, 256, 0, stream>>>(
        t1, wpk + (size_t)(2 * l + 1) * HD * HD, l2b[l], agg, stats, N);
    bn_finalize<<<1, 128, 0, stream>>>(stats, bn_g[l], bn_b[l], ssc, N);
    bn_apply<<<elBlocks, 256, 0, stream>>>(agg, ssc, h, n4);
  }

  pool_zero<<<(BGR * HD + 255) / 256, 256, 0, stream>>>(hg, counts);
  pool_scatter<<<(int)(((long long)N * 32 + 255) / 256), 256, 0, stream>>>(h, batch, hg, counts, N);
  pool_finalize<<<(BGR * HD + 255) / 256, 256, 0, stream>>>(hg, counts);

  phys_enc<<<BGR, 128, 0, stream>>>(u, ph_w, ph_b, ph_lng, ph_lnb, hphys);
  fusion<<<BGR, 256, 0, stream>>>(hg, HD, hphys, HD, f1_w, f1_b, f1_lng, f1_lnb, hf1);
  fusion<<<BGR, 256, 0, stream>>>(hf1, MD, hf1 + HD, MD, f2_w, f2_b, f2_lng, f2_lnb, hf2);
  heads<<<BGR, 256, 0, stream>>>(hf2, th_w, th_b, z_w, z_b, (float*)d_out);
}
